// LSTM_46677704573211
// MI455X (gfx1250) — matrix operand order repaired and checked by reading, not compiled, not run on hardware
//
#include <hip/hip_runtime.h>
#include <hip/hip_bf16.h>

// Problem constants
#define S_  512
#define B_  64
#define D_  1024
#define H_  1024
#define L_  2
#define G4_ 4096          // 4*H
#define EPSV 1e-5f

typedef __attribute__((ext_vector_type(16))) __bf16 v16bf;
typedef __attribute__((ext_vector_type(8)))  __bf16 v8bf;
typedef __attribute__((ext_vector_type(8)))  float  v8f;

// ---------------------------------------------------------------------------
// gfx1250 async global->LDS copy (ASYNCcnt path, bypasses VGPRs).
// VDST = wave-relative LDS byte offset; VADDR = 64-bit global address.
// LDS offset from generic shared pointer: flat->LDS mapping uses addr[31:0].
// ---------------------------------------------------------------------------
__device__ __forceinline__ void async_copy16(const void* lds_ptr, const void* gptr) {
    unsigned int lds_off = (unsigned int)(uintptr_t)lds_ptr;
    asm volatile("global_load_async_to_lds_b128 %0, %1, off"
                 :: "v"(lds_off), "v"((unsigned long long)(uintptr_t)gptr)
                 : "memory");
}
__device__ __forceinline__ void wait_async0() {
#if __has_builtin(__builtin_amdgcn_s_wait_asynccnt)
    __builtin_amdgcn_s_wait_asynccnt(0);
#else
    asm volatile("s_wait_asynccnt 0x0" ::: "memory");
#endif
}

// ---------------------------------------------------------------------------
// WMMA fragment loaders (layouts per CDNA5 ISA §7.12.2, wave32)
// A: 16x32 bf16 row-major tile in LDS, pitch 32 elements.
//   lane<16 : row=lane,    frag[0:7]=K 0..7,  frag[8:15]=K 16..23
//   lane>=16: row=lane-16, frag[0:7]=K 8..15, frag[8:15]=K 24..31
// ---------------------------------------------------------------------------
__device__ __forceinline__ v16bf load_a_frag(const __bf16* sA, int lane) {
    int r  = lane & 15;
    int kb = (lane < 16) ? 0 : 8;
    v8bf lo = *(const v8bf*)(sA + r * 32 + kb);
    v8bf hi = *(const v8bf*)(sA + r * 32 + kb + 16);
    return __builtin_shufflevector(lo, hi, 0,1,2,3,4,5,6,7,8,9,10,11,12,13,14,15);
}
// B: 32x16 bf16, tile staged N-major in LDS (pitch 32 K-elements per N row).
__device__ __forceinline__ v16bf load_b_frag(const __bf16* sBn, int lane) {
    int n   = lane & 15;
    int kb2 = (lane < 16) ? 0 : 8;
    v8bf lo = *(const v8bf*)(sBn + n * 32 + kb2);
    v8bf hi = *(const v8bf*)(sBn + n * 32 + kb2 + 16);
    return __builtin_shufflevector(lo, hi, 0,1,2,3,4,5,6,7,8,9,10,11,12,13,14,15);
}

// ---------------------------------------------------------------------------
// One-shot: transpose + convert weights to bf16, N-major (4H x K)
// ---------------------------------------------------------------------------
__global__ __launch_bounds__(256)
void k_wt(const float* __restrict__ wx, const float* __restrict__ wh,
          __bf16* __restrict__ wxT, __bf16* __restrict__ whT) {
    size_t idx = (size_t)blockIdx.x * 256 + threadIdx.x;   // over L*4H*D
    size_t per = (size_t)G4_ * D_;
    int    l   = (int)(idx / per);
    size_t rem = idx % per;
    int    n   = (int)(rem / D_);
    int    k   = (int)(rem % D_);
    size_t src = (size_t)l * D_ * G4_ + (size_t)k * G4_ + n;
    wxT[idx] = (__bf16)wx[src];
    whT[idx] = (__bf16)wh[src];
}

// ---------------------------------------------------------------------------
// Big parallel GEMM: xg[m,n] = sum_k A[m,k]*BT[n,k] + bias[n]
// A f32 (M=S*B x 1024), BT bf16 (4096 x 1024, N-major).
// WG tile 128x128, 8 waves, wave tile 32x64, BK=32, double-buffered LDS.
// B tiles staged via async-to-LDS; A staged via VALU (needs f32->bf16 cvt).
// ---------------------------------------------------------------------------
__global__ __launch_bounds__(256)
void k_gemm_xg(const float* __restrict__ A, const __bf16* __restrict__ BT,
               const float* __restrict__ bias, float* __restrict__ C) {
    __shared__ __align__(32) __bf16 sA[2][128 * 32];
    __shared__ __align__(32) __bf16 sB[2][128 * 32];
    const int t = threadIdx.x, lane = t & 31, wave = t >> 5;
    const int wm = wave & 3;          // M slab of 32
    const int wn = wave >> 2;         // N slab of 64
    const int rowBase = blockIdx.y * 128;
    const int colBase = blockIdx.x * 128;

    auto stageA = [&](int buf, int kb) {        // 128x32 f32 -> bf16 (VALU path)
#pragma unroll
        for (int it = 0; it < 4; ++it) {
            int slot = t + it * 256;
            int r = slot >> 3, kq = (slot & 7) * 4;
            float4 v = *(const float4*)(A + (size_t)(rowBase + r) * 1024 + kb + kq);
            __bf16* d = &sA[buf][r * 32 + kq];
            d[0] = (__bf16)v.x; d[1] = (__bf16)v.y;
            d[2] = (__bf16)v.z; d[3] = (__bf16)v.w;
        }
    };
    auto stageB = [&](int buf, int kb) {        // 128x32 bf16 (async path)
#pragma unroll
        for (int it = 0; it < 2; ++it) {
            int slot = t + it * 256;
            int r = slot >> 2, kq = (slot & 3) * 8;
            async_copy16(&sB[buf][r * 32 + kq],
                         BT + (size_t)(colBase + r) * 1024 + kb + kq);
        }
    };

    v8f zero = {};
    v8f acc[2][4];
#pragma unroll
    for (int mi = 0; mi < 2; ++mi)
#pragma unroll
        for (int ni = 0; ni < 4; ++ni) acc[mi][ni] = zero;

    stageB(0, 0);
    stageA(0, 0);

    for (int kb = 0, itn = 0; kb < 1024; kb += 32, ++itn) {
        const int cur = itn & 1;
        wait_async0();                 // my async B copies for buf[cur] done
        __syncthreads();               // everyone's copies/stores for buf[cur] done
        if (kb + 32 < 1024) {          // overlap next tile with current compute
            stageB(cur ^ 1, kb + 32);
            stageA(cur ^ 1, kb + 32);
        }
        if (kb + 64 < 1024) {          // global_prefetch_b8 of A stream
            __builtin_prefetch(A + (size_t)(rowBase + (t >> 1)) * 1024 + kb + 64, 0, 3);
        }

        v16bf af[2], bfr[4];
#pragma unroll
        for (int mi = 0; mi < 2; ++mi)
            af[mi] = load_a_frag(&sA[cur][(wm * 32 + mi * 16) * 32], lane);
#pragma unroll
        for (int ni = 0; ni < 4; ++ni)
            bfr[ni] = load_b_frag(&sB[cur][(wn * 64 + ni * 16) * 32], lane);
#pragma unroll
        for (int mi = 0; mi < 2; ++mi)
#pragma unroll
            for (int ni = 0; ni < 4; ++ni)
                acc[mi][ni] = __builtin_amdgcn_wmma_f32_16x16x32_bf16(
                    false, af[mi], false, bfr[ni], (short)0, acc[mi][ni], false, false);
    }

    const int ro = (lane >> 4) * 8;
    const int co = lane & 15;
#pragma unroll
    for (int mi = 0; mi < 2; ++mi)
#pragma unroll
        for (int ni = 0; ni < 4; ++ni)
#pragma unroll
            for (int j = 0; j < 8; ++j) {
                int row = rowBase + wm * 32 + mi * 16 + ro + j;
                int col = colBase + wn * 64 + ni * 16 + co;
                C[(size_t)row * G4_ + col] = acc[mi][ni][j] + bias[col];
            }
}

// ---------------------------------------------------------------------------
// Per-step recurrent GEMM: G[b,n] = xg_s[b,n] + sum_k Hb[b,k]*BT[n,k]
// WG tile 64x128 (full batch M), grid 32 WGs, wave tile 16x64, BK=32.
// Both A (bf16 h) and B tiles staged via async-to-LDS, double-buffered.
// ---------------------------------------------------------------------------
__global__ __launch_bounds__(256)
void k_gemm_step(const __bf16* __restrict__ Hb, const __bf16* __restrict__ BT,
                 const float* __restrict__ xg_s, float* __restrict__ G) {
    __shared__ __align__(32) __bf16 sA[2][64 * 32];
    __shared__ __align__(32) __bf16 sB[2][128 * 32];
    const int t = threadIdx.x, lane = t & 31, wave = t >> 5;
    const int wm = wave & 3;          // M slab of 16
    const int wn = wave >> 2;         // N slab of 64
    const int colBase = blockIdx.x * 128;

    auto stage = [&](int buf, int kb) {
        {   // A: 64x32 bf16, 256 x 16B async copies
            int r = t >> 2, kq = (t & 3) * 8;
            async_copy16(&sA[buf][r * 32 + kq],
                         Hb + (size_t)r * 1024 + kb + kq);
        }
#pragma unroll
        for (int it = 0; it < 2; ++it) {   // B: 128x32 bf16
            int slot = t + it * 256;
            int r = slot >> 2, kq = (slot & 3) * 8;
            async_copy16(&sB[buf][r * 32 + kq],
                         BT + (size_t)(colBase + r) * 1024 + kb + kq);
        }
    };

    v8f zero = {};
    v8f acc[4];
#pragma unroll
    for (int ni = 0; ni < 4; ++ni) acc[ni] = zero;

    stage(0, 0);

    for (int kb = 0, itn = 0; kb < 1024; kb += 32, ++itn) {
        const int cur = itn & 1;
        wait_async0();
        __syncthreads();
        if (kb + 32 < 1024) stage(cur ^ 1, kb + 32);

        v16bf af = load_a_frag(&sA[cur][(wm * 16) * 32], lane);
        v16bf bfr[4];
#pragma unroll
        for (int ni = 0; ni < 4; ++ni)
            bfr[ni] = load_b_frag(&sB[cur][(wn * 64 + ni * 16) * 32], lane);
#pragma unroll
        for (int ni = 0; ni < 4; ++ni)
            acc[ni] = __builtin_amdgcn_wmma_f32_16x16x32_bf16(
                false, af, false, bfr[ni], (short)0, acc[ni], false, false);
    }

    const int ro = (lane >> 4) * 8;
    const int co = lane & 15;
#pragma unroll
    for (int ni = 0; ni < 4; ++ni)
#pragma unroll
        for (int j = 0; j < 8; ++j) {
            int row = wm * 16 + ro + j;
            int col = colBase + wn * 64 + ni * 16 + co;
            size_t off = (size_t)row * G4_ + col;
            G[off] = acc[ni][j] + xg_s[off];
        }
}

// ---------------------------------------------------------------------------
// Per-step pointwise: LayerNorm per gate group + activations + state update.
// One block per batch row. wave32 shfl_xor reductions + single barrier.
// ---------------------------------------------------------------------------
__device__ __forceinline__ float sigmf(float x) { return 1.0f / (1.0f + __expf(-x)); }

__global__ __launch_bounds__(256)
void k_point(const float* __restrict__ G, const float* __restrict__ gamma,
             const float* __restrict__ beta, float* __restrict__ cst,
             float* __restrict__ ys, __bf16* __restrict__ hbf,
             float* __restrict__ hfin) {
    __shared__ float s1[4][8], s2[4][8];
    const int b = blockIdx.x, t = threadIdx.x;
    const int lane = t & 31, wave = t >> 5;
    const float* g = G + (size_t)b * G4_;

    float vals[4][4];
#pragma unroll
    for (int gi = 0; gi < 4; ++gi) {
        float sum = 0.f, ssq = 0.f;
#pragma unroll
        for (int j = 0; j < 4; ++j) {
            float x = g[gi * 1024 + t + j * 256];
            vals[gi][j] = x;
            sum += x; ssq += x * x;
        }
#pragma unroll
        for (int m = 16; m > 0; m >>= 1) {   // wave32 butterfly reduction
            sum += __shfl_xor(sum, m, 32);
            ssq += __shfl_xor(ssq, m, 32);
        }
        if (lane == 0) { s1[gi][wave] = sum; s2[gi][wave] = ssq; }
    }
    __syncthreads();

    float mu[4], rs[4];
#pragma unroll
    for (int gi = 0; gi < 4; ++gi) {
        float a = 0.f, q = 0.f;
#pragma unroll
        for (int w = 0; w < 8; ++w) { a += s1[gi][w]; q += s2[gi][w]; }
        float m = a * (1.0f / 1024.0f);
        float v = q * (1.0f / 1024.0f) - m * m;
        mu[gi] = m;
        rs[gi] = rsqrtf(v + EPSV);
    }
#pragma unroll
    for (int gi = 0; gi < 4; ++gi)
#pragma unroll
        for (int j = 0; j < 4; ++j) {
            int hidx = t + j * 256;
            vals[gi][j] = (vals[gi][j] - mu[gi]) * rs[gi] * gamma[gi * 1024 + hidx]
                        + beta[gi * 1024 + hidx];
        }
#pragma unroll
    for (int j = 0; j < 4; ++j) {
        int hidx = t + j * 256;
        size_t off = (size_t)b * 1024 + hidx;
        float gi = vals[0][j], gf = vals[1][j], go = vals[2][j], gu = vals[3][j];
        float c = sigmf(gf) * cst[off] + sigmf(gi) * tanhf(gu);
        float h = sigmf(go) * tanhf(c);
        cst[off]  = c;
        ys[off]   = h;
        hbf[off]  = (__bf16)h;
        hfin[off] = h;
    }
}

// Init per-layer state: h0 -> bf16, c0 -> f32 working state.
__global__ __launch_bounds__(256)
void k_init(const float* __restrict__ h0l, const float* __restrict__ c0l,
            __bf16* __restrict__ hbf, float* __restrict__ cst) {
    int i = blockIdx.x * 256 + threadIdx.x;   // B*H
    hbf[i] = (__bf16)h0l[i];
    cst[i] = c0l[i];
}

// Final hs/cs copy into d_out tail.
__global__ __launch_bounds__(256)
void k_fin(const float* __restrict__ hfin, const float* __restrict__ cst,
           float* __restrict__ outp) {
    int i = blockIdx.x * 256 + threadIdx.x;   // L*B*H
    outp[i] = hfin[i];
    outp[L_ * B_ * H_ + i] = cst[i];
}

// ---------------------------------------------------------------------------
extern "C" void kernel_launch(void* const* d_in, const int* in_sizes, int n_in,
                              void* d_out, int out_size, void* d_ws, size_t ws_size,
                              hipStream_t stream) {
    const float* x_in  = (const float*)d_in[0];   // (S,B,D)
    const float* h0    = (const float*)d_in[1];   // (L,B,H)
    const float* c0    = (const float*)d_in[2];   // (L,B,H)
    const float* wx    = (const float*)d_in[3];   // (L,D,4H)
    const float* wh    = (const float*)d_in[4];   // (L,H,4H)
    const float* bias  = (const float*)d_in[5];   // (L,4H)
    const float* gamma = (const float*)d_in[6];   // (L,4,H)
    const float* beta  = (const float*)d_in[7];   // (L,4,H)

    char* ws = (char*)d_ws;
    const size_t XG_B   = (size_t)S_ * B_ * G4_ * 4;       // f32 gates precompute
    const size_t YS0_B  = (size_t)S_ * B_ * H_ * 4;        // layer-0 output
    const size_t WT_B   = (size_t)L_ * G4_ * D_ * 2;       // bf16 weights (each)
    const size_t GB_B   = (size_t)B_ * G4_ * 4;            // step gates
    const size_t HBF_B  = (size_t)B_ * H_ * 2;             // bf16 h state
    const size_t HF_B   = (size_t)L_ * B_ * H_ * 4;

    float*  xg   = (float*) (ws);
    float*  ys0  = (float*) (ws + XG_B);
    __bf16* wxT  = (__bf16*)(ws + XG_B + YS0_B);
    __bf16* whT  = (__bf16*)(ws + XG_B + YS0_B + WT_B);
    float*  gbuf = (float*) (ws + XG_B + YS0_B + 2 * WT_B);
    __bf16* hbf  = (__bf16*)(ws + XG_B + YS0_B + 2 * WT_B + GB_B);
    float*  hfin = (float*) (ws + XG_B + YS0_B + 2 * WT_B + GB_B + HBF_B);
    float*  cst  = (float*) (ws + XG_B + YS0_B + 2 * WT_B + GB_B + HBF_B + HF_B);

    // 1) Weights: transpose + f32->bf16 (once per call).
    k_wt<<<(L_ * G4_ * D_) / 256, 256, 0, stream>>>(wx, wh, wxT, whT);

    for (int l = 0; l < L_; ++l) {
        const float* Ain = (l == 0) ? x_in : ys0;
        // 2) xg = x @ wx + bias : (32768 x 1024) x (1024 x 4096)
        k_gemm_xg<<<dim3(G4_ / 128, (S_ * B_) / 128), 256, 0, stream>>>(
            Ain, wxT + (size_t)l * G4_ * D_, bias + (size_t)l * G4_, xg);

        k_init<<<(B_ * H_) / 256, 256, 0, stream>>>(
            h0 + (size_t)l * B_ * H_, c0 + (size_t)l * B_ * H_,
            hbf, cst + (size_t)l * B_ * H_);

        float* ys = (l == 0) ? ys0 : (float*)d_out;
        for (int s = 0; s < S_; ++s) {
            // 3) gates = xg[s] + h @ wh
            k_gemm_step<<<G4_ / 128, 256, 0, stream>>>(
                hbf, whT + (size_t)l * G4_ * H_,
                xg + (size_t)s * B_ * G4_, gbuf);
            // 4) LayerNorm + activations + (h,c) update
            k_point<<<B_, 256, 0, stream>>>(
                gbuf, gamma + (size_t)l * 4 * H_, beta + (size_t)l * 4 * H_,
                cst + (size_t)l * B_ * H_, ys + (size_t)s * B_ * H_,
                hbf, hfin + (size_t)l * B_ * H_);
        }
    }

    // 5) hs / cs tail of d_out
    k_fin<<<(L_ * B_ * H_) / 256, 256, 0, stream>>>(
        hfin, cst, (float*)d_out + (size_t)S_ * B_ * H_);
}
